// CAFormer_80951543595511
// MI455X (gfx1250) — matrix operand order repaired and checked by reading, not compiled, not run on hardware
//
#include <hip/hip_runtime.h>
#include <stdint.h>

// ---------------- problem constants ----------------
#define DIMC   256
#define NHEAD  8
#define HEADC  32
#define BATCHN 8
#define HH     56
#define WW     56
#define HW     3136          // 56*56
#define NELEM  ((size_t)BATCHN * DIMC * HW)   // 6,422,528

// ---------------- WMMA types ----------------
typedef __attribute__((ext_vector_type(16))) __bf16 bf16x16;
typedef __attribute__((ext_vector_type(8)))  float  f32x8;
typedef __attribute__((ext_vector_type(4)))  int    i32x4;

union FragAB {
    bf16x16 v;
    uint4   u[2];
};

// ---------------- bf16 helpers (bit-level, RNE) ----------------
__device__ __forceinline__ float bf2f(unsigned short u) {
    union { float f; uint32_t i; } c;
    c.i = ((uint32_t)u) << 16;
    return c.f;
}
__device__ __forceinline__ unsigned short f2bf(float f) {
    union { float f; uint32_t i; } c;
    c.f = f;
    uint32_t u = c.i;
    uint32_t r = (u + 0x7FFFu + ((u >> 16) & 1u)) >> 16;
    return (unsigned short)r;
}

// ---------------- gfx1250 async global->LDS copy (16B per lane) -------
#if __has_builtin(__builtin_amdgcn_global_load_async_to_lds_b128)
#define HAVE_ASYNC_LDS 1
#endif

__device__ __forceinline__ void async_cp16(const unsigned short* g, unsigned short* l) {
#ifdef HAVE_ASYNC_LDS
    __builtin_amdgcn_global_load_async_to_lds_b128(
        (__attribute__((address_space(1))) i32x4*)g,
        (__attribute__((address_space(3))) i32x4*)l, 0, 0);
#else
    *(uint4*)l = *(const uint4*)g;
#endif
}
__device__ __forceinline__ void async_wait_all() {
#ifdef HAVE_ASYNC_LDS
#if __has_builtin(__builtin_amdgcn_s_wait_asynccnt)
    __builtin_amdgcn_s_wait_asynccnt(0);
#else
    asm volatile("s_wait_asynccnt 0" ::: "memory");
#endif
#endif
}

// ---------------- f32 -> bf16 (weights, no transpose) -----------------
__global__ void k_f32_to_bf16(const float* __restrict__ s,
                              unsigned short* __restrict__ d, int n) {
    int i = blockIdx.x * 256 + threadIdx.x;
    if (i < n) d[i] = f2bf(s[i]);
}

// ---------------- x (NCHW f32) -> xb_t (NHWC bf16) --------------------
// Reads coalesced along pos; scattered 2B writes are absorbed by 192MB L2.
__global__ void k_x_to_bf16_t(const float* __restrict__ x,
                              unsigned short* __restrict__ xt) {
    size_t i = (size_t)blockIdx.x * 256 + threadIdx.x;   // pos fastest
    int pos = (int)(i % HW);
    size_t t = i / HW;
    int c = (int)(t % DIMC);
    int b = (int)(t / DIMC);
    xt[((size_t)b * HW + pos) * DIMC + c] = f2bf(x[i]);
}

// ---------------- depthwise 3x3 qk conv -> q/k NHWC bf16 --------------
__global__ void k_qk_dw(const float* __restrict__ x,
                        const float* __restrict__ w,   // (512,1,3,3)
                        const float* __restrict__ bqk, // (512)
                        unsigned short* __restrict__ qbuf,
                        unsigned short* __restrict__ kbuf) {
    int idx = blockIdx.x * 256 + threadIdx.x;  // over 8*512*3136
    int pos = idx % HW;
    int t   = idx / HW;
    int ch  = t % (2 * DIMC);
    int b   = t / (2 * DIMC);
    if (b >= BATCHN) return;
    int cin = ch & (DIMC - 1);
    int y = pos / WW, xx = pos % WW;
    const float* xin = x + ((size_t)b * DIMC + cin) * HW;
    const float* wp  = w + ch * 9;
    float acc = bqk[ch];
#pragma unroll
    for (int p = 0; p < 9; ++p) {
        int yy = y + p / 3 - 1;
        int xc = xx + p % 3 - 1;
        float v = 0.f;
        if (yy >= 0 && yy < HH && xc >= 0 && xc < WW) v = xin[yy * WW + xc];
        acc += wp[p] * v;
    }
    unsigned short o = f2bf(acc);
    size_t oi = ((size_t)b * HW + pos) * DIMC + cin;   // NHWC
    if (ch < DIMC) qbuf[oi] = o; else kbuf[oi] = o;
}

// ---------------- WMMA GEMM core (double-buffered, async-LDS) ---------
// A = weights bf16 [256][256] row-major (K contiguous)
// B = activations bf16 NHWC [pos][256] (K contiguous) -> both tiles are
//     plain 64B-row copies; no LDS transpose scatter needed.
#define BM 128
#define BN 128
#define BK 32

__device__ __forceinline__ void load_tiles_async(
    const unsigned short* __restrict__ W,
    const unsigned short* __restrict__ Bx,
    int m0, int n0, int kt,
    unsigned short* As, unsigned short* Bs) {
    const int tid = threadIdx.x;
#pragma unroll
    for (int c = 0; c < 2; ++c) {
        int chunk = tid * 2 + c;       // 0..511
        int row   = chunk >> 2;        // 0..127
        int cc    = chunk & 3;         // 0..3 (8 ushorts)
        async_cp16(&W[(size_t)(m0 + row) * DIMC + kt + cc * 8],
                   &As[row * BK + cc * 8]);
    }
#pragma unroll
    for (int c = 0; c < 2; ++c) {
        int chunk = tid * 2 + c;
        int row   = chunk >> 2;
        int cc    = chunk & 3;
        int pos   = n0 + row;
        if (pos >= HW) pos = 0;        // clamp: junk columns discarded at store
        async_cp16(&Bx[(size_t)pos * DIMC + kt + cc * 8],
                   &Bs[row * BK + cc * 8]);
    }
}

__device__ __forceinline__ void gemm_core(
    const unsigned short* __restrict__ W,
    const unsigned short* __restrict__ Bx,
    int m0, int n0,
    unsigned short (*As)[BM * BK], unsigned short (*Bs)[BN * BK],
    f32x8 acc[4][2]) {
    const int tid  = threadIdx.x;
    const int lane = tid & 31;
    const int wid  = tid >> 5;
    const int wr   = wid >> 2;   // 0..1
    const int wc   = wid & 3;    // 0..3
    const int lm   = lane & 15;
    const int hi   = lane >> 4;  // 0 or 1

#pragma unroll
    for (int i = 0; i < 4; ++i)
#pragma unroll
        for (int j = 0; j < 2; ++j)
            acc[i][j] = (f32x8){0.f, 0.f, 0.f, 0.f, 0.f, 0.f, 0.f, 0.f};

    // prologue: stage first K tile
    load_tiles_async(W, Bx, m0, n0, 0, As[0], Bs[0]);
    async_wait_all();
    __syncthreads();

    int buf = 0;
    for (int kt = 0; kt < DIMC; kt += BK, buf ^= 1) {
        // overlap: fetch next tile while computing this one
        if (kt + BK < DIMC)
            load_tiles_async(W, Bx, m0, n0, kt + BK, As[buf ^ 1], Bs[buf ^ 1]);

        // fragments (ISA 16-bit A/B per-lane layouts)
        FragAB afr[4], bfr[2];
#pragma unroll
        for (int ms = 0; ms < 4; ++ms) {
            int rb = (wr * 64 + ms * 16 + lm) * BK;
            afr[ms].u[0] = *(const uint4*)&As[buf][rb + (hi ? 8 : 0)];
            afr[ms].u[1] = *(const uint4*)&As[buf][rb + (hi ? 24 : 16)];
        }
#pragma unroll
        for (int ns = 0; ns < 2; ++ns) {
            int rb = (wc * 32 + ns * 16 + lm) * BK;
            bfr[ns].u[0] = *(const uint4*)&Bs[buf][rb + (hi ? 8 : 0)];
            bfr[ns].u[1] = *(const uint4*)&Bs[buf][rb + (hi ? 24 : 16)];
        }

#pragma unroll
        for (int ms = 0; ms < 4; ++ms)
#pragma unroll
            for (int ns = 0; ns < 2; ++ns)
                acc[ms][ns] = __builtin_amdgcn_wmma_f32_16x16x32_bf16(
                    false, afr[ms].v, false, bfr[ns].v,
                    (short)0, acc[ms][ns], false, false);

        async_wait_all();   // next buffer resident (own wave)
        __syncthreads();    // ... and for every wave in the block
    }
}

// ---------------- v = v_w @ x + v_b  -> NHWC bf16 ---------------------
__global__ void __launch_bounds__(256)
k_v_gemm(const unsigned short* __restrict__ Wv,
         const unsigned short* __restrict__ xbt,
         const float* __restrict__ vb,
         unsigned short* __restrict__ vout) {
    __shared__ unsigned short As[2][BM * BK];
    __shared__ unsigned short Bs[2][BN * BK];
    int b  = blockIdx.z;
    int m0 = blockIdx.y * BM;
    int n0 = blockIdx.x * BN;
    const unsigned short* Bx = xbt + (size_t)b * HW * DIMC;
    f32x8 acc[4][2];
    gemm_core(Wv, Bx, m0, n0, As, Bs, acc);

    const int lane = threadIdx.x & 31;
    const int wid  = threadIdx.x >> 5;
    const int wr = wid >> 2, wc = wid & 3;
    const int lm = lane & 15, hi = lane >> 4;
    unsigned short* outT = vout + (size_t)b * HW * DIMC;
#pragma unroll
    for (int ms = 0; ms < 4; ++ms)
#pragma unroll
        for (int ns = 0; ns < 2; ++ns) {
            int n  = n0 + wc * 32 + ns * 16 + lm;
            int mb = m0 + wr * 64 + ms * 16 + hi * 8;   // 8 consecutive m
            if (n < HW) {
                union { uint4 v; unsigned short s[8]; } pk;
#pragma unroll
                for (int r = 0; r < 8; ++r)
                    pk.s[r] = f2bf(acc[ms][ns][r] + vb[mb + r]);
                *(uint4*)&outT[(size_t)n * DIMC + mb] = pk.v;  // 16B store
            }
        }
}

// ---------------- per-pixel channel attention (1 wave / pixel) --------
// NHWC layout: lane = channel within head -> all global accesses are
// 64B-contiguous across the wave's 32 lanes.
__global__ void __launch_bounds__(256)
k_attn(const unsigned short* __restrict__ qb,
       const unsigned short* __restrict__ kb,
       const unsigned short* __restrict__ vb,
       unsigned short* __restrict__ qkv) {
    __shared__ float ksh[8][HEADC][9];
    __shared__ float vsh[8][HEADC];
    const int w    = threadIdx.x >> 5;
    const int lane = threadIdx.x & 31;
    int gw  = blockIdx.x * 8 + w;         // 0 .. 64*3136-1
    int pos = gw % HW;
    int bh  = gw / HW;                    // 0..63
    int b = bh >> 3, head = bh & 7;
    int y = pos / WW, x = pos % WW;
    int ch = head * HEADC + lane;
    size_t rowb = (size_t)b * HW;

    float qr[9];
#pragma unroll
    for (int p = 0; p < 9; ++p) {
        int yy = y + p / 3 - 1;
        int xx = x + p % 3 - 1;
        bool ok = (yy >= 0 && yy < HH && xx >= 0 && xx < WW);
        size_t a = (rowb + (size_t)(yy * WW + xx)) * DIMC + ch;
        qr[p]           = ok ? bf2f(qb[a]) : 0.f;
        ksh[w][lane][p] = ok ? bf2f(kb[a]) : 0.f;
    }
    vsh[w][lane] = bf2f(vb[(rowb + pos) * DIMC + ch]);
    __syncthreads();

    float s[HEADC];
    float mx = -1e30f;
#pragma unroll
    for (int j = 0; j < HEADC; ++j) {
        float t = 0.f;
#pragma unroll
        for (int p = 0; p < 9; ++p) t += qr[p] * ksh[w][j][p];
        s[j] = t;
        mx = fmaxf(mx, t);
    }
    float sum = 0.f, av = 0.f;
#pragma unroll
    for (int j = 0; j < HEADC; ++j) {
        float e = __expf(s[j] - mx);
        sum += e;
        av  += e * vsh[w][j];
    }
    qkv[(rowb + pos) * DIMC + ch] = f2bf(av / sum);
}

// ---------------- proj GEMM + bias + BN + SiLU + residual -> NCHW f32 -
__global__ void __launch_bounds__(256)
k_proj_gemm(const unsigned short* __restrict__ Wp,
            const unsigned short* __restrict__ qkvb,
            const float* __restrict__ pb,
            const float* __restrict__ gamma,
            const float* __restrict__ beta,
            const float* __restrict__ mean,
            const float* __restrict__ var,
            const float* __restrict__ xres,
            float* __restrict__ out) {
    __shared__ unsigned short As[2][BM * BK];
    __shared__ unsigned short Bs[2][BN * BK];
    int b  = blockIdx.z;
    int m0 = blockIdx.y * BM;
    int n0 = blockIdx.x * BN;
    const unsigned short* Bx = qkvb + (size_t)b * HW * DIMC;
    f32x8 acc[4][2];
    gemm_core(Wp, Bx, m0, n0, As, Bs, acc);

    const int lane = threadIdx.x & 31;
    const int wid  = threadIdx.x >> 5;
    const int wr = wid >> 2, wc = wid & 3;
    const int lm = lane & 15, hi = lane >> 4;
#pragma unroll
    for (int ms = 0; ms < 4; ++ms)
#pragma unroll
        for (int ns = 0; ns < 2; ++ns) {
            int mb = m0 + wr * 64 + ms * 16 + hi * 8;
            int n  = n0 + wc * 32 + ns * 16 + lm;
#pragma unroll
            for (int r = 0; r < 8; ++r) {
                int mm = mb + r;
                if (n < HW) {
                    float a   = acc[ms][ns][r] + pb[mm];
                    float inv = gamma[mm] * rsqrtf(var[mm] + 1e-5f);
                    float xn  = (a - mean[mm]) * inv + beta[mm];
                    float sg  = 1.f / (1.f + __expf(-xn));
                    size_t oi = ((size_t)b * DIMC + mm) * HW + n;
                    out[oi] = xn * sg + xres[oi];
                }
            }
        }
}

// ---------------- launcher ----------------
extern "C" void kernel_launch(void* const* d_in, const int* in_sizes, int n_in,
                              void* d_out, int out_size, void* d_ws, size_t ws_size,
                              hipStream_t stream) {
    const float* x      = (const float*)d_in[0];
    const float* qk_w   = (const float*)d_in[1];
    const float* qk_b   = (const float*)d_in[2];
    const float* v_w    = (const float*)d_in[3];
    const float* v_b    = (const float*)d_in[4];
    const float* proj_w = (const float*)d_in[5];
    const float* proj_b = (const float*)d_in[6];
    const float* gamma  = (const float*)d_in[7];
    const float* beta   = (const float*)d_in[8];
    const float* mean   = (const float*)d_in[9];
    const float* var    = (const float*)d_in[10];
    float* out = (float*)d_out;

    char* ws = (char*)d_ws;
    unsigned short* xbt  = (unsigned short*)ws; ws += NELEM * 2;  // NHWC bf16
    unsigned short* wvb  = (unsigned short*)ws; ws += (size_t)DIMC * DIMC * 2;
    unsigned short* wpb  = (unsigned short*)ws; ws += (size_t)DIMC * DIMC * 2;
    unsigned short* qbuf = (unsigned short*)ws; ws += NELEM * 2;  // NHWC bf16
    unsigned short* kbuf = (unsigned short*)ws; ws += NELEM * 2;
    unsigned short* vbuf = (unsigned short*)ws; ws += NELEM * 2;
    unsigned short* qkvb = (unsigned short*)ws; ws += NELEM * 2;

    // 1) conversions (x transposed to NHWC)
    k_x_to_bf16_t<<<(unsigned)((NELEM + 255) / 256), 256, 0, stream>>>(x, xbt);
    k_f32_to_bf16<<<(DIMC * DIMC + 255) / 256, 256, 0, stream>>>(v_w, wvb, DIMC * DIMC);
    k_f32_to_bf16<<<(DIMC * DIMC + 255) / 256, 256, 0, stream>>>(proj_w, wpb, DIMC * DIMC);

    // 2) depthwise qk conv (NHWC bf16 out)
    k_qk_dw<<<(unsigned)((size_t)BATCHN * 2 * DIMC * HW / 256), 256, 0, stream>>>(
        x, qk_w, qk_b, qbuf, kbuf);

    // 3) v = v_w @ x + v_b   (WMMA, async-LDS double buffered)
    dim3 gg((HW + BN - 1) / BN, DIMC / BM, BATCHN);   // 25 x 2 x 8
    k_v_gemm<<<gg, 256, 0, stream>>>(wvb, xbt, v_b, vbuf);

    // 4) per-pixel channel attention
    k_attn<<<(unsigned)((size_t)NHEAD * BATCHN * HW / 8), 256, 0, stream>>>(
        qbuf, kbuf, vbuf, qkvb);

    // 5) proj GEMM + BN + SiLU + residual  (WMMA)
    k_proj_gemm<<<gg, 256, 0, stream>>>(wpb, qkvb, proj_b, gamma, beta, mean, var, x, out);
}
